// DAGNN_24979529793772
// MI455X (gfx1250) — compile-verified, matrix operand order checked by reading, not run on hardware
//
#include <hip/hip_runtime.h>
#include <hip/hip_bf16.h>

// Problem constants (from reference): B=64, N=128, H=512, L=2
#define B_   64
#define N_   128
#define H_   512
#define L_   2
#define H3_  1536

typedef __attribute__((ext_vector_type(16))) __bf16 v16bf;
typedef __attribute__((ext_vector_type(8)))  __bf16 v8bf;
typedef __attribute__((ext_vector_type(8)))  float  v8f;

static __device__ __forceinline__ int laneid() { return threadIdx.x & 31; }

// ---- WMMA fragment loaders (ISA 7.12.2 layouts, wave32) ----
// A: 16x32 bf16 from f32 source, converted with native cvt (v_cvt_pk_bf16_f32).
// Lane l<16 holds row m0+l, K = k0+{0..7} and k0+16+{0..7};
// lane l+16 holds row m0+l, K = k0+8+{0..7} and k0+24+{0..7}.
static __device__ __forceinline__ v16bf load_a_f32(const float* __restrict__ A,
                                                   int lda, int m0, int k0) {
    int l = laneid();
    const float* base = A + (size_t)(m0 + (l & 15)) * lda + k0 + ((l & 16) ? 8 : 0);
    float4 x0 = *(const float4*)(base);        // K+0..3
    float4 x1 = *(const float4*)(base + 4);    // K+4..7
    float4 x2 = *(const float4*)(base + 16);   // K+16..19
    float4 x3 = *(const float4*)(base + 20);   // K+20..23
    v16bf a;
    a[0]  = (__bf16)x0.x; a[1]  = (__bf16)x0.y; a[2]  = (__bf16)x0.z; a[3]  = (__bf16)x0.w;
    a[4]  = (__bf16)x1.x; a[5]  = (__bf16)x1.y; a[6]  = (__bf16)x1.z; a[7]  = (__bf16)x1.w;
    a[8]  = (__bf16)x2.x; a[9]  = (__bf16)x2.y; a[10] = (__bf16)x2.z; a[11] = (__bf16)x2.w;
    a[12] = (__bf16)x3.x; a[13] = (__bf16)x3.y; a[14] = (__bf16)x3.z; a[15] = (__bf16)x3.w;
    return a;
}

// B: 32x16 bf16 with B[k][n] = W[n][k]; W is bf16 row-major (ldw along k).
// Two 128-bit loads per lane, symmetric layout to A with N in place of M.
static __device__ __forceinline__ v16bf load_b_w(const unsigned short* __restrict__ W,
                                                 int ldw, int n0, int k0) {
    int l = laneid();
    const unsigned short* base =
        W + (size_t)(n0 + (l & 15)) * ldw + k0 + ((l & 16) ? 8 : 0);
    v8bf lo = *(const v8bf*)(base);        // K+0..7
    v8bf hi = *(const v8bf*)(base + 16);   // K+16..23
    return __builtin_shufflevector(lo, hi, 0, 1, 2, 3, 4, 5, 6, 7,
                                          8, 9, 10, 11, 12, 13, 14, 15);
}

// D/C 16x16 f32: VGPR j -> row m0+j (lanes 0-15) / m0+8+j (lanes 16-31), col n0+(lane&15).
template<bool BIAS, bool ACC>
static __device__ __forceinline__ void store_d(float* __restrict__ O, int ldo,
                                               int m0, int n0, v8f d,
                                               const float* __restrict__ bias) {
    int l     = laneid();
    int col   = n0 + (l & 15);
    int rbase = m0 + ((l & 16) ? 8 : 0);
    float bv  = BIAS ? bias[col] : 0.0f;
#pragma unroll
    for (int j = 0; j < 8; ++j) {
        float* p = O + (size_t)(rbase + j) * ldo + col;
        float  v = d[j] + bv;
        if (ACC) v += *p;
        *p = v;
    }
}

#define WMMA_BF16(a, b, c) \
    __builtin_amdgcn_wmma_f32_16x16x32_bf16(false, (a), false, (b), (short)0, (c), false, false)

// One wave computes a 16x64 output strip: 4 N-tiles share each A fragment,
// so every A load feeds 4 back-to-back v_wmma (high matrix-pipe occupancy).
template<bool BIAS, bool ACC>
static __device__ __forceinline__ void gemm_wave_16x64(
    const float* __restrict__ A, int lda,
    const unsigned short* __restrict__ W, int ldw,
    const float* __restrict__ bias, float* __restrict__ O, int ldo,
    int m0, int n0, int K) {
    v8f c0 = {}, c1 = {}, c2 = {}, c3 = {};
    for (int k0 = 0; k0 < K; k0 += 32) {
        if (k0 + 32 < K)
            __builtin_prefetch(W + (size_t)(n0 + (laneid() & 15)) * ldw + k0 + 32, 0, 1);
        v16bf a  = load_a_f32(A, lda, m0, k0);
        v16bf b0 = load_b_w(W, ldw, n0,      k0);
        v16bf b1 = load_b_w(W, ldw, n0 + 16, k0);
        v16bf b2 = load_b_w(W, ldw, n0 + 32, k0);
        v16bf b3 = load_b_w(W, ldw, n0 + 48, k0);
        c0 = WMMA_BF16(a, b0, c0);
        c1 = WMMA_BF16(a, b1, c1);
        c2 = WMMA_BF16(a, b2, c2);
        c3 = WMMA_BF16(a, b3, c3);
    }
    store_d<BIAS, ACC>(O, ldo, m0, n0,      c0, bias);
    store_d<BIAS, ACC>(O, ldo, m0, n0 + 16, c1, bias);
    store_d<BIAS, ACC>(O, ldo, m0, n0 + 32, c2, bias);
    store_d<BIAS, ACC>(O, ldo, m0, n0 + 48, c3, bias);
}

// =================== utility kernels ===================

__global__ void k_cvt_bf16(const float* __restrict__ src, unsigned short* __restrict__ dst, int n) {
    int i = blockIdx.x * blockDim.x + threadIdx.x;
    int s = gridDim.x * blockDim.x;
    for (; i < n; i += s) {
        __bf16 h = (__bf16)src[i];
        dst[i] = __builtin_bit_cast(unsigned short, h);
    }
}

__global__ void k_zero_f32(float* __restrict__ dst, int n) {
    int i = blockIdx.x * blockDim.x + threadIdx.x;
    int s = gridDim.x * blockDim.x;
    for (; i < n; i += s) dst[i] = 0.0f;
}

// qw[row] = dot(X[row,:H], v)  — one block per row
__global__ void k_dotv(const float* __restrict__ X, const float* __restrict__ v,
                       float* __restrict__ out) {
    __shared__ float red[256];
    int row = blockIdx.x, t = threadIdx.x;
    const float* x = X + (size_t)row * H_;
    float s = 0.0f;
    for (int h = t; h < H_; h += 256) s += x[h] * v[h];
    red[t] = s; __syncthreads();
    for (int off = 128; off > 0; off >>= 1) {
        if (t < off) red[t] += red[t + off];
        __syncthreads();
    }
    if (t == 0) out[row] = red[0];
}

// =================== WMMA GEMM kernels ===================

// out[(b,n),h] (+)= Hpart[(b,n),:512] . embW_bf[h, :512] (+bias). M=8192,N=512,K=512.
// 4096 wave-strips = 512 (m) x 8 (n-groups of 64); 8 waves / block.
template<bool BIAS, bool ACC>
__global__ void k_emb_gemm(const float* __restrict__ A, const unsigned short* __restrict__ W,
                           const float* __restrict__ bias, float* __restrict__ O) {
    int t  = blockIdx.x * 8 + (threadIdx.x >> 5);
    int m0 = (t & 511) * 16;
    int n0 = (t >> 9) * 64;
    gemm_wave_16x64<BIAS, ACC>(A, H_, W, H3_, bias, O, H_, m0, n0, H_);
}

// M = A0 @ W0^T + A1 @ W1^T   (64x512, K=512 each). 32 strips = 4 (m) x 8 (n-groups).
__global__ void k_m_gemm(const float* __restrict__ A0, const float* __restrict__ A1,
                         const unsigned short* __restrict__ W0, const unsigned short* __restrict__ W1,
                         float* __restrict__ M) {
    int t  = blockIdx.x * 8 + (threadIdx.x >> 5);
    int m0 = (t & 3) * 16;
    int n0 = (t >> 2) * 64;
    v8f c0 = {}, c1 = {}, c2 = {}, c3 = {};
    for (int k0 = 0; k0 < H_; k0 += 32) {
        v16bf a  = load_a_f32(A0, H_, m0, k0);
        v16bf b0 = load_b_w(W0, H_, n0,      k0);
        v16bf b1 = load_b_w(W0, H_, n0 + 16, k0);
        v16bf b2 = load_b_w(W0, H_, n0 + 32, k0);
        v16bf b3 = load_b_w(W0, H_, n0 + 48, k0);
        c0 = WMMA_BF16(a, b0, c0);
        c1 = WMMA_BF16(a, b1, c1);
        c2 = WMMA_BF16(a, b2, c2);
        c3 = WMMA_BF16(a, b3, c3);
    }
    for (int k0 = 0; k0 < H_; k0 += 32) {
        v16bf a  = load_a_f32(A1, H_, m0, k0);
        v16bf b0 = load_b_w(W1, H_, n0,      k0);
        v16bf b1 = load_b_w(W1, H_, n0 + 16, k0);
        v16bf b2 = load_b_w(W1, H_, n0 + 32, k0);
        v16bf b3 = load_b_w(W1, H_, n0 + 48, k0);
        c0 = WMMA_BF16(a, b0, c0);
        c1 = WMMA_BF16(a, b1, c1);
        c2 = WMMA_BF16(a, b2, c2);
        c3 = WMMA_BF16(a, b3, c3);
    }
    store_d<false, false>(M, H_, m0, n0,      c0, nullptr);
    store_d<false, false>(M, H_, m0, n0 + 16, c1, nullptr);
    store_d<false, false>(M, H_, m0, n0 + 32, c2, nullptr);
    store_d<false, false>(M, H_, m0, n0 + 48, c3, nullptr);
}

// Four GRU gate GEMMs (blockIdx.y selects): each (64 x 1536), K=512.
//   0: q @ cWih^T + cbih     1: M @ cWhh^T + cbhh
//   2: M @ pWih^T + pbih     3: q @ pWhh^T + pbhh
// Per which: 96 strips = 4 (m) x 24 (n-groups of 64); grid (12, 4) x 256.
__global__ void k_gates_gemm(const float* __restrict__ Hq,   // Hcur + i*H_, lda = N_*H_
                             const float* __restrict__ Mb,   // lda = H_
                             const unsigned short* __restrict__ cWih,
                             const unsigned short* __restrict__ cWhh,
                             const unsigned short* __restrict__ pWih,
                             const unsigned short* __restrict__ pWhh,
                             const float* __restrict__ cbih, const float* __restrict__ cbhh,
                             const float* __restrict__ pbih, const float* __restrict__ pbhh,
                             float* __restrict__ gates) {
    int which = blockIdx.y;
    const float* A; int lda; const unsigned short* W; const float* bias;
    if      (which == 0) { A = Hq; lda = N_ * H_; W = cWih; bias = cbih; }
    else if (which == 1) { A = Mb; lda = H_;      W = cWhh; bias = cbhh; }
    else if (which == 2) { A = Mb; lda = H_;      W = pWih; bias = pbih; }
    else                 { A = Hq; lda = N_ * H_; W = pWhh; bias = pbhh; }
    float* O = gates + (size_t)which * B_ * H3_;

    int t  = blockIdx.x * 8 + (threadIdx.x >> 5);
    int m0 = (t & 3) * 16;
    int n0 = (t >> 2) * 64;
    gemm_wave_16x64<true, false>(A, lda, W, H_, bias, O, H3_, m0, n0, H_);
}

// =================== scan step kernels ===================

// Attention + softmax + weighted reductions A0 = sum_n w*s*H1[n], A1 = sum_n w*o*H1[n]
__global__ void k_att(int i, const float* __restrict__ qw, const float* __restrict__ hk,
                      const int* __restrict__ adj, const int* __restrict__ smask,
                      const int* __restrict__ omask, const float* __restrict__ H1,
                      const float* __restrict__ attb,
                      float* __restrict__ A0, float* __restrict__ A1) {
    __shared__ float sw[N_], ow[N_], red[N_];
    int b = blockIdx.x, t = threadIdx.x;   // 128 threads, one per n
    int base = b * N_ * N_ + i * N_;
    float ab = attb[0];
    int aval = (t < i) ? adj[base + t] : 0;
    float alpha = (aval != 0) ? (qw[b * N_ + i] + hk[b * N_ + t] + ab) : -1e30f;
    red[t] = alpha; __syncthreads();
    for (int off = 64; off > 0; off >>= 1) {
        if (t < off) red[t] = fmaxf(red[t], red[t + off]);
        __syncthreads();
    }
    float mx = red[0]; __syncthreads();
    float e = __expf(alpha - mx);
    red[t] = e; __syncthreads();
    for (int off = 64; off > 0; off >>= 1) {
        if (t < off) red[t] += red[t + off];
        __syncthreads();
    }
    float w = e / red[0];
    sw[t] = w * (float)smask[base + t];
    ow[t] = w * (float)omask[base + t];
    __syncthreads();

    const float* Hb = H1 + (size_t)b * N_ * H_;
    for (int h = t; h < H_; h += N_) {
        float a0 = 0.0f, a1 = 0.0f;
        for (int n = 0; n < i; ++n) {   // w == 0 exactly for n >= i (exp(-inf))
            float v = Hb[(size_t)n * H_ + h];
            a0 += sw[n] * v;
            a1 += ow[n] * v;
        }
        A0[b * H_ + h] = a0;
        A1[b * H_ + h] = a1;
    }
}

// GRU elementwise: H1[:,i] = Ci + Pi; hk[:,i] = H1[:,i] . wk
__global__ void k_combine(int i, const float* __restrict__ Hcur, const float* __restrict__ Mb,
                          const float* __restrict__ gates, const float* __restrict__ wk,
                          float* __restrict__ H1, float* __restrict__ hk) {
    __shared__ float red[H_];
    int b = blockIdx.x, h = threadIdx.x;   // 512 threads
    const float* gic = gates + 0 * (size_t)B_ * H3_ + (size_t)b * H3_;
    const float* ghc = gates + 1 * (size_t)B_ * H3_ + (size_t)b * H3_;
    const float* gip = gates + 2 * (size_t)B_ * H3_ + (size_t)b * H3_;
    const float* ghp = gates + 3 * (size_t)B_ * H3_ + (size_t)b * H3_;
    float q = Hcur[(size_t)b * N_ * H_ + (size_t)i * H_ + h];
    float m = Mb[b * H_ + h];

    float rc = 1.0f / (1.0f + __expf(-(gic[h] + ghc[h])));
    float zc = 1.0f / (1.0f + __expf(-(gic[H_ + h] + ghc[H_ + h])));
    float nc = tanhf(gic[2 * H_ + h] + rc * ghc[2 * H_ + h]);
    float Ci = (1.0f - zc) * nc + zc * m;

    float rp = 1.0f / (1.0f + __expf(-(gip[h] + ghp[h])));
    float zp = 1.0f / (1.0f + __expf(-(gip[H_ + h] + ghp[H_ + h])));
    float np = tanhf(gip[2 * H_ + h] + rp * ghp[2 * H_ + h]);
    float Pi = (1.0f - zp) * np + zp * q;

    float val = Ci + Pi;
    H1[(size_t)b * N_ * H_ + (size_t)i * H_ + h] = val;

    red[h] = val * wk[h]; __syncthreads();
    for (int off = 256; off > 0; off >>= 1) {
        if (h < off) red[h] += red[h + off];
        __syncthreads();
    }
    if (h == 0) hk[b * N_ + i] = red[0];
}

// =================== host orchestration ===================

extern "C" void kernel_launch(void* const* d_in, const int* in_sizes, int n_in,
                              void* d_out, int out_size, void* d_ws, size_t ws_size,
                              hipStream_t stream) {
    (void)in_sizes; (void)n_in; (void)out_size; (void)ws_size;
    const float* feats = (const float*)d_in[0];
    const int*   adj   = (const int*)d_in[1];
    const int*   smask = (const int*)d_in[2];
    const int*   omask = (const int*)d_in[3];
    const float* cWih  = (const float*)d_in[4];
    const float* cWhh  = (const float*)d_in[5];
    const float* cbih  = (const float*)d_in[6];
    const float* cbhh  = (const float*)d_in[7];
    const float* pWih  = (const float*)d_in[8];
    const float* pWhh  = (const float*)d_in[9];
    const float* pbih  = (const float*)d_in[10];
    const float* pbhh  = (const float*)d_in[11];
    const float* attW  = (const float*)d_in[12];
    const float* attb  = (const float*)d_in[13];
    const float* Wr0   = (const float*)d_in[14];
    const float* Wr1   = (const float*)d_in[15];
    const float* embW  = (const float*)d_in[16];
    const float* embb  = (const float*)d_in[17];
    float* out = (float*)d_out;

    // ---- workspace carve-up (~52 MB) ----
    char* cur = (char*)d_ws;
    auto carve = [&](size_t bytes) -> void* {
        void* p = cur; cur += (bytes + 255) & ~(size_t)255; return p;
    };
    const size_t WG = (size_t)L_ * H3_ * H_;   // gate weight elems per tensor
    const size_t WR = (size_t)L_ * H_ * H_;
    unsigned short* cWih_bf = (unsigned short*)carve(WG * 2);
    unsigned short* cWhh_bf = (unsigned short*)carve(WG * 2);
    unsigned short* pWih_bf = (unsigned short*)carve(WG * 2);
    unsigned short* pWhh_bf = (unsigned short*)carve(WG * 2);
    unsigned short* W0_bf   = (unsigned short*)carve(WR * 2);
    unsigned short* W1_bf   = (unsigned short*)carve(WR * 2);
    unsigned short* emb_bf  = (unsigned short*)carve((size_t)H_ * H3_ * 2);
    float* H1a   = (float*)carve((size_t)B_ * N_ * H_ * 4);
    float* H1b   = (float*)carve((size_t)B_ * N_ * H_ * 4);
    float* qw    = (float*)carve((size_t)B_ * N_ * 4);
    float* hk    = (float*)carve((size_t)B_ * N_ * 4);
    float* A0    = (float*)carve((size_t)B_ * H_ * 4);
    float* A1    = (float*)carve((size_t)B_ * H_ * 4);
    float* Mbuf  = (float*)carve((size_t)B_ * H_ * 4);
    float* gates = (float*)carve((size_t)4 * B_ * H3_ * 4);

    // ---- one-time weight conversion to bf16 ----
    k_cvt_bf16<<<512, 256, 0, stream>>>(cWih, cWih_bf, (int)WG);
    k_cvt_bf16<<<512, 256, 0, stream>>>(cWhh, cWhh_bf, (int)WG);
    k_cvt_bf16<<<512, 256, 0, stream>>>(pWih, pWih_bf, (int)WG);
    k_cvt_bf16<<<512, 256, 0, stream>>>(pWhh, pWhh_bf, (int)WG);
    k_cvt_bf16<<<512, 256, 0, stream>>>(Wr0, W0_bf, (int)WR);
    k_cvt_bf16<<<512, 256, 0, stream>>>(Wr1, W1_bf, (int)WR);
    k_cvt_bf16<<<512, 256, 0, stream>>>(embW, emb_bf, H_ * H3_);

    // ---- output part 0: feats @ embW[:, 0:H]^T + emb_b ----
    k_emb_gemm<true, false><<<512, 256, 0, stream>>>(feats, emb_bf + 0, embb, out);

    for (int l = 0; l < L_; ++l) {
        const float* Hcur = (l == 0) ? feats : H1a;
        float* H1 = (l == 0) ? H1a : H1b;
        const unsigned short* lcWih = cWih_bf + (size_t)l * H3_ * H_;
        const unsigned short* lcWhh = cWhh_bf + (size_t)l * H3_ * H_;
        const unsigned short* lpWih = pWih_bf + (size_t)l * H3_ * H_;
        const unsigned short* lpWhh = pWhh_bf + (size_t)l * H3_ * H_;
        const unsigned short* lW0   = W0_bf + (size_t)l * H_ * H_;
        const unsigned short* lW1   = W1_bf + (size_t)l * H_ * H_;
        const float* lcbih = cbih + (size_t)l * H3_;
        const float* lcbhh = cbhh + (size_t)l * H3_;
        const float* lpbih = pbih + (size_t)l * H3_;
        const float* lpbhh = pbhh + (size_t)l * H3_;
        const float* wq = attW + (size_t)l * 2 * H_;
        const float* wk = wq + H_;
        const float* ab = attb + l;

        // per-layer init: zero state, precompute qw = Hcur . wq
        k_zero_f32<<<2048, 256, 0, stream>>>(H1, B_ * N_ * H_);
        k_zero_f32<<<64, 256, 0, stream>>>(Mbuf, B_ * H_);
        k_zero_f32<<<32, 256, 0, stream>>>(hk, B_ * N_);
        k_dotv<<<B_ * N_, 256, 0, stream>>>(Hcur, wq, qw);

        // step 0: M = 0 makes the two GRU cells reduce exactly to the reference H0
        {
            dim3 g(12, 4);
            k_gates_gemm<<<g, 256, 0, stream>>>(Hcur /*+0*H_*/, Mbuf,
                                                lcWih, lcWhh, lpWih, lpWhh,
                                                lcbih, lcbhh, lpbih, lpbhh, gates);
            k_combine<<<B_, H_, 0, stream>>>(0, Hcur, Mbuf, gates, wk, H1, hk);
        }

        // sequential scan i = 1..N-1
        for (int i = 1; i < N_; ++i) {
            k_att<<<B_, N_, 0, stream>>>(i, qw, hk, adj, smask, omask, H1, ab, A0, A1);
            k_m_gemm<<<4, 256, 0, stream>>>(A0, A1, lW0, lW1, Mbuf);
            dim3 g(12, 4);
            k_gates_gemm<<<g, 256, 0, stream>>>(Hcur + (size_t)i * H_, Mbuf,
                                                lcWih, lcWhh, lpWih, lpWhh,
                                                lcbih, lcbhh, lpbih, lpbhh, gates);
            k_combine<<<B_, H_, 0, stream>>>(i, Hcur, Mbuf, gates, wk, H1, hk);
        }
    }

    // ---- output parts 1,2: accumulate layer outputs through emb ----
    k_emb_gemm<false, true><<<512, 256, 0, stream>>>(H1a, emb_bf + H_,     nullptr, out);
    k_emb_gemm<false, true><<<512, 256, 0, stream>>>(H1b, emb_bf + 2 * H_, nullptr, out);
}
